// Sparse_Graph_Attention_Layer_62130996903989
// MI455X (gfx1250) — matrix operand order repaired and checked by reading, not compiled, not run on hardware
//
#include <hip/hip_runtime.h>
#include <hip/hip_bf16.h>

typedef _Float16 v16h __attribute__((ext_vector_type(16)));
typedef float    v8f  __attribute__((ext_vector_type(8)));

#define GN     50000
#define GDEG   32
#define GK     256     // IN_DIM
#define GD     128     // OUT_DIM
#define GALPHA 0.2f

#define MROWS  80      // rows per block (50000 = 625 * 80)
#define MT     5       // 16-row tiles per wave
#define LXP    260     // padded f32 row stride for x tile  (bank advance 4)
#define LWP    264     // padded f16 row stride for W^T tile (bank advance 4)

// ---------------------------------------------------------------------------
// Kernel 1: wh = x @ W  via v_wmma_f32_16x16x32_f16.
// Block = 256 threads = 8 waves; block covers 80 rows x 128 cols.
//  - x tile (80x256 f32, 80 KB) staged via global_load_async_to_lds_b128
//    (ASYNCcnt), overlapped with W staging, completed with s_wait_asynccnt.
//  - W staged once per block into LDS, TRANSPOSED + f16 (lds_wt[n][k]):
//    a B fragment is two 16-byte runs/lane (2x ds_load_b128) and is
//    reused across 5 WMMAs (5 M-tiles per wave, 40 WMMAs/wave total).
// Fragment layouts per CDNA5 ISA 7.12.2 (wave32):
//   A 16x32 f16 : lanes 0-15 row=lane, K {0..7,16..23}; lanes 16-31 K {8..15,24..31}
//   B 32x16 f16 : lanes 0-15 K {0..7,16..23} (col=lane&15), lanes 16-31 K {8..15,24..31}
//   C/D 16x16 f32: VGPR r -> M=r / r+8, N=lane&15
// ---------------------------------------------------------------------------
__global__ __launch_bounds__(256)
void gat_gemm_wh(const float* __restrict__ x, const float* __restrict__ w,
                 float* __restrict__ wh) {
  __shared__ float    lds_x[MROWS][LXP];  // ~83 KB
  __shared__ _Float16 lds_wt[GD][LWP];    // ~66 KB, W transposed [n][k]

  const int tid  = threadIdx.x;
  const int lane = tid & 31;
  const int wave = tid >> 5;
  const int row0 = blockIdx.x * MROWS;

  // ---- Stage x tile via async global->LDS: 80*256 f32 = 5120 x 16B chunks ----
  #pragma unroll 5
  for (int i = 0; i < 20; ++i) {
    const int c  = tid + 256 * i;         // chunk id
    const int m  = c >> 6;                // 64 chunks (256 f32) per row
    const int k0 = (c & 63) << 2;         // float offset in row
    const unsigned ldsa = (unsigned)(size_t)&lds_x[m][k0];
    const float* g = x + (size_t)(row0 + m) * GK + k0;
    asm volatile("global_load_async_to_lds_b128 %0, %1, off"
                 :: "v"(ldsa), "v"(g) : "memory");
  }

  // ---- Stage W (256x128 f32) -> LDS transposed f16 (overlaps async x) ----
  #pragma unroll 4
  for (int i = 0; i < 32; ++i) {
    const int flat = tid + 256 * i;       // float4 index over 256x128
    const int k  = flat >> 5;             // 0..255
    const int n0 = (flat & 31) << 2;      // 0..124
    const float4 v = *(const float4*)(w + (size_t)k * GD + n0);
    lds_wt[n0 + 0][k] = (_Float16)v.x;
    lds_wt[n0 + 1][k] = (_Float16)v.y;
    lds_wt[n0 + 2][k] = (_Float16)v.z;
    lds_wt[n0 + 3][k] = (_Float16)v.w;
  }

  asm volatile("s_wait_asynccnt 0x0" ::: "memory");
  __syncthreads();

  // ---- 8 K-steps x 5 M-tiles of WMMA, pure LDS inner loop ----
  const int col0 = wave * 16;
  const int a_m  = lane & 15;
  const int a_kb = (lane >> 4) << 3;      // 0 or 8
  const int b_n  = col0 + (lane & 15);
  const int b_kb = (lane >> 4) << 3;      // 0 or 8

  v8f acc[MT] = {};
  #pragma unroll
  for (int kt = 0; kt < GK; kt += 32) {
    v16h b;
    {
      const _Float16* __restrict__ q = &lds_wt[b_n][kt + b_kb];
      #pragma unroll
      for (int e = 0; e < 8; ++e) { b[e] = q[e]; b[8 + e] = q[16 + e]; }
    }
    #pragma unroll
    for (int t = 0; t < MT; ++t) {
      v16h a;
      const float* __restrict__ p0 = &lds_x[t * 16 + a_m][kt + a_kb];
      const float* __restrict__ p1 = &lds_x[t * 16 + a_m][kt + 16 + a_kb];
      #pragma unroll
      for (int e = 0; e < 8; ++e) {
        a[e]     = (_Float16)p0[e];
        a[8 + e] = (_Float16)p1[e];
      }
      acc[t] = __builtin_amdgcn_wmma_f32_16x16x32_f16(
          /*neg_a=*/false, a, /*neg_b=*/false, b,
          /*c_mod=*/(short)0, acc[t], /*reuse_a=*/false, /*reuse_b=*/false);
    }
  }

  const int c_mb = (lane >> 4) << 3;      // 0 or 8
  const int c_n  = lane & 15;
  #pragma unroll
  for (int t = 0; t < MT; ++t) {
    #pragma unroll
    for (int r = 0; r < 8; ++r) {
      wh[(size_t)(row0 + t * 16 + c_mb + r) * GD + col0 + c_n] = acc[t][r];
    }
  }
}

// ---------------------------------------------------------------------------
// Kernel 2: s1[i] = dot(wh[i], a1); s2[i] = dot(wh[i], a2).
// One wave32 per row; 4 elements per lane; xor-shuffle reduction.
// ---------------------------------------------------------------------------
__global__ __launch_bounds__(256)
void gat_scores(const float* __restrict__ wh, const float* __restrict__ av,
                float* __restrict__ s1, float* __restrict__ s2) {
  const int lane = threadIdx.x & 31;
  const int wave = threadIdx.x >> 5;
  const int row  = blockIdx.x * 8 + wave;
  if (row >= GN) return;

  const float* __restrict__ r = wh + (size_t)row * GD;
  float p1 = 0.f, p2 = 0.f;
  #pragma unroll
  for (int t = 0; t < 4; ++t) {
    const int c = lane * 4 + t;
    const float v = r[c];
    p1 += v * av[c];
    p2 += v * av[GD + c];
  }
  #pragma unroll
  for (int off = 16; off >= 1; off >>= 1) {
    p1 += __shfl_xor(p1, off, 32);
    p2 += __shfl_xor(p2, off, 32);
  }
  if (lane == 0) { s1[row] = p1; s2[row] = p2; }
}

// ---------------------------------------------------------------------------
// Kernel 3: softmax-weighted neighbor aggregation + ELU.
// One wave32 per node; lane owns 4 output dims (float4). wh (25.6 MB) stays
// resident in MI455X's 192 MB L2, so the ~819 MB of gathers are on-chip.
// Edges for node i are slots [i*32, (i+1)*32) since row = repeat(arange(N),32).
// ---------------------------------------------------------------------------
__global__ __launch_bounds__(256)
void gat_aggregate(const float* __restrict__ wh, const int* __restrict__ col,
                   const float* __restrict__ s1, const float* __restrict__ s2,
                   float* __restrict__ out) {
  const int lane = threadIdx.x & 31;
  const int wave = threadIdx.x >> 5;
  const int node = blockIdx.x * 8 + wave;
  if (node >= GN) return;

  const float si = s1[node];
  const int* __restrict__ ecol = col + (size_t)node * GDEG;

  float ax = 0.f, ay = 0.f, az = 0.f, aw = 0.f, den = 0.f;
  #pragma unroll 4
  for (int j = 0; j < GDEG; ++j) {
    const int c  = ecol[j];
    const float sc = si + s2[c];
    const float lr = sc > 0.f ? sc : GALPHA * sc;
    const float e  = __expf(-lr);
    const float4 v = *(const float4*)(wh + (size_t)c * GD + lane * 4);
    ax += e * v.x; ay += e * v.y; az += e * v.z; aw += e * v.w;
    den += e;
  }
  const float inv = 1.f / den;
  float4 o;
  o.x = ax * inv; o.y = ay * inv; o.z = az * inv; o.w = aw * inv;
  o.x = o.x > 0.f ? o.x : (__expf(o.x) - 1.f);
  o.y = o.y > 0.f ? o.y : (__expf(o.y) - 1.f);
  o.z = o.z > 0.f ? o.z : (__expf(o.z) - 1.f);
  o.w = o.w > 0.f ? o.w : (__expf(o.w) - 1.f);
  *(float4*)(out + (size_t)node * GD + lane * 4) = o;
}

// ---------------------------------------------------------------------------
extern "C" void kernel_launch(void* const* d_in, const int* in_sizes, int n_in,
                              void* d_out, int out_size, void* d_ws, size_t ws_size,
                              hipStream_t stream) {
  const float* x    = (const float*)d_in[0];
  const int*   edge = (const int*)d_in[1];   // [2, E] flattened; col = edge + E
  const float* w    = (const float*)d_in[2];
  const float* av   = (const float*)d_in[3]; // [1, 2*GD]
  float*       out  = (float*)d_out;

  const int E = GN * GDEG;
  float* wh = (float*)d_ws;                  // N*GD floats = 25.6 MB
  float* s1 = wh + (size_t)GN * GD;          // N floats
  float* s2 = s1 + GN;                       // N floats
  const int* col = edge + E;

  (void)in_sizes; (void)n_in; (void)out_size; (void)ws_size;

  hipLaunchKernelGGL(gat_gemm_wh,   dim3(GN / MROWS),   dim3(256), 0, stream, x, w, wh);
  hipLaunchKernelGGL(gat_scores,    dim3((GN + 7) / 8), dim3(256), 0, stream, wh, av, s1, s2);
  hipLaunchKernelGGL(gat_aggregate, dim3((GN + 7) / 8), dim3(256), 0, stream, wh, col, s1, s2, out);
}
